// TranCFormer_10196252361525
// MI455X (gfx1250) — matrix operand order repaired and checked by reading, not compiled, not run on hardware
//
#include <hip/hip_runtime.h>
#include <hip/hip_bf16.h>
#include <stddef.h>
#include <stdint.h>

// ---------------- problem constants ----------------
#define B_    16
#define C_    256
#define H_    80
#define W_    80
#define SP    (H_*W_)        // 6400 pixels per batch
#define NPIX  (B_*SP)        // 102400 total pixels
#define QKVO  160            // qkv output channels
#define CATC  384            // concat channels (128 attn + 256 hid)
#define EPSV  1e-5f

// ---------------- vector types ----------------
typedef __attribute__((ext_vector_type(8)))  float          f32x8;
typedef __attribute__((ext_vector_type(8)))  unsigned short u16x8;
typedef __attribute__((ext_vector_type(16))) unsigned short u16x16;
typedef __attribute__((ext_vector_type(16))) __bf16         bf16x16;
typedef __attribute__((ext_vector_type(4)))  unsigned int   u32x4;
typedef __attribute__((ext_vector_type(8)))  unsigned int   u32x8;

union FragU { u16x8 h[2]; u16x16 v; };

__device__ __forceinline__ unsigned short f2bf(float f) {
    unsigned int u = __float_as_uint(f);
    u += 0x7FFFu + ((u >> 16) & 1u);          // round-to-nearest-even
    return (unsigned short)(u >> 16);
}
__device__ __forceinline__ float siluf(float v) {
    return v / (1.f + __expf(-v));
}
__device__ __forceinline__ f32x8 wmma_bf16(u16x16 a, u16x16 b, f32x8 c) {
    union Cv { u16x16 u; bf16x16 bf; };
    Cv A; A.u = a;
    Cv B; B.u = b;
    // 8 args: (neg_a, A, neg_b, B, c_mod, C, reuse_a, reuse_b)
    return __builtin_amdgcn_wmma_f32_16x16x32_bf16(false, A.bf, false, B.bf,
                                                   (short)0, c, false, false);
}

// ---------------- setup kernels ----------------

// NCHW f32 -> NHWC bf16 (LDS tiled transpose of [256][6400] per batch)
__global__ void k_x_to_nhwc(const float* __restrict__ x, unsigned short* __restrict__ xb) {
    __shared__ float tile[32][33];
    int bz = blockIdx.z;
    int c0 = blockIdx.y * 32;
    int s0 = blockIdx.x * 32;
    int tx = threadIdx.x & 31, ty = threadIdx.x >> 5;   // 8 rows of 32
    for (int r = ty; r < 32; r += 8)
        tile[r][tx] = x[((size_t)(bz * C_ + c0 + r)) * SP + s0 + tx];
    __syncthreads();
    for (int r = ty; r < 32; r += 8) {
        int s = s0 + r, c = c0 + tx;
        xb[((size_t)bz * SP + s) * C_ + c] = f2bf(tile[tx][r]);
    }
}

// fold all three BN layers: inv = g*rsqrt(v+eps), bias = b - m*inv
__global__ void k_fold_bn(const float* qg, const float* qb, const float* qm, const float* qv,
                          const float* hg, const float* hb, const float* hm, const float* hv,
                          const float* og, const float* ob, const float* om, const float* ov,
                          float* q_inv, float* q_bias, float* h_inv, float* h_bias,
                          float* o_inv, float* o_bias) {
    int t = threadIdx.x;
    if (t < C_) {
        float hi = hg[t] * rsqrtf(hv[t] + EPSV);
        h_inv[t] = hi;  h_bias[t] = hb[t] - hm[t] * hi;
        float oi = og[t] * rsqrtf(ov[t] + EPSV);
        o_inv[t] = oi;  o_bias[t] = ob[t] - om[t] * oi;
    }
    if (t < QKVO) {
        float qi = qg[t] * rsqrtf(qv[t] + EPSV);
        q_inv[t] = qi;  q_bias[t] = qb[t] - qm[t] * qi;
    }
}

// hid_w OIHW f32 -> bf16 K-panel layout Bp[(k/16)*256 + o][k%16], k = (ky*3+kx)*256 + i
__global__ void k_pack_hid(const float* __restrict__ w, unsigned short* __restrict__ bp) {
    int idx = blockIdx.x * 256 + threadIdx.x;       // 256*256*9 total
    int o = idx / 2304, r = idx % 2304;
    int i = r / 9, t = r % 9;
    int k = t * C_ + i;
    bp[(((size_t)(k >> 4)) * C_ + o) * 16 + (k & 15)] = f2bf(w[idx]);
}

// out_w [o][384] f32 -> bf16 panels Bp2[(k/16)*256 + o][k%16]
__global__ void k_pack_out(const float* __restrict__ w, unsigned short* __restrict__ bp) {
    int idx = blockIdx.x * 256 + threadIdx.x;       // 256*384 total
    int o = idx / CATC, k = idx % CATC;
    bp[(((size_t)(k >> 4)) * C_ + o) * 16 + (k & 15)] = f2bf(w[idx]);
}

// ---------------- fused qkv grouped-conv + per-pixel attention ----------------
__global__ void k_qkv_attn(const float* __restrict__ x, const float* __restrict__ w,
                           const float* __restrict__ inv, const float* __restrict__ bias,
                           unsigned short* __restrict__ cat) {
    __shared__ float qb[16 * QKVO];
    int t  = threadIdx.x;
    int px = t >> 4;               // 0..15
    int c0 = (t & 15) * 10;        // 0,10,..,150 (each span lies in one group of 20)
    int p  = blockIdx.x * 16 + px;
    int b  = p / SP, s = p % SP, y = s / W_, x0 = s % W_;
    int g  = c0 / 20;

    float acc[10];
#pragma unroll
    for (int i = 0; i < 10; i++) acc[i] = 0.f;

    for (int ky = 0; ky < 3; ky++) {
        int yy = y + ky - 1;
        if (yy < 0 || yy >= H_) continue;
        for (int kx = 0; kx < 3; kx++) {
            int xc = x0 + kx - 1;
            if (xc < 0 || xc >= W_) continue;
            const float* xp = x + ((size_t)(b * C_ + g * 32)) * SP + yy * W_ + xc;
            const float* wp = w + (size_t)c0 * 288 + ky * 3 + kx;   // + ci*9 + oc*288
            for (int ci = 0; ci < 32; ci++) {
                float xv = xp[(size_t)ci * SP];
                const float* wq = wp + ci * 9;
#pragma unroll
                for (int oc = 0; oc < 10; oc++) acc[oc] += xv * wq[oc * 288];
            }
        }
    }
#pragma unroll
    for (int oc = 0; oc < 10; oc++) {
        int c = c0 + oc;
        qb[px * QKVO + c] = siluf(acc[oc] * inv[c] + bias[c]);
    }
    __syncthreads();

    // attention: row i of pixel px
    int i = t & 15;
    const float* qk = &qb[px * QKVO];
    float qi = qk[i];
    float sc[16], mx = -1e30f;
#pragma unroll
    for (int j = 0; j < 16; j++) { sc[j] = qi * qk[16 + j]; mx = fmaxf(mx, sc[j]); }
    float den = 0.f;
#pragma unroll
    for (int j = 0; j < 16; j++) { sc[j] = __expf(sc[j] - mx); den += sc[j]; }
    float rd = 1.f / den;
    float yv[8];
#pragma unroll
    for (int d = 0; d < 8; d++) yv[d] = 0.f;
#pragma unroll
    for (int j = 0; j < 16; j++) {
        float a = sc[j];
        const float* vp = &qk[32 + j * 8];
#pragma unroll
        for (int d = 0; d < 8; d++) yv[d] += a * vp[d];
    }
    unsigned short* op = cat + (size_t)p * CATC + i * 8;
#pragma unroll
    for (int d = 0; d < 8; d++) op[d] = f2bf(yv[d] * rd);
}

// ---------------- hid 3x3 conv 256->256 : implicit GEMM via WMMA ----------------
// block tile: M=64 pixels x N=128 out-channels, 8 waves, 4 C-tiles/wave.
// Per tap: async-DMA 64x256 bf16 im2col slab into LDS, 8 K-steps of 32, 4 WMMA each.
__global__ void k_hid_conv(const unsigned short* __restrict__ xb,
                           const unsigned short* __restrict__ bp,
                           const float* __restrict__ inv, const float* __restrict__ bias,
                           unsigned short* __restrict__ cat) {
    __shared__ __align__(16) unsigned short As[64 * C_];   // 32 KB
    int tid  = threadIdx.x;
    int wv   = tid >> 5, lane = tid & 31;
    int h    = lane >> 4, r = lane & 15;
    int mi   = wv & 3;           // M sub-tile 0..3
    int np   = wv >> 2;          // N half 0..1 -> cols np*64 + t*16
    int m0   = blockIdx.x * 64;
    int n0   = blockIdx.y * 128;

    f32x8 acc[4];
#pragma unroll
    for (int t2 = 0; t2 < 4; t2++) acc[t2] = (f32x8){0,0,0,0,0,0,0,0};

    for (int tap = 0; tap < 9; tap++) {
        int ky = tap / 3, kx = tap % 3;
        __syncthreads();
        // stage im2col slab for this tap: 64 px * 256 ch bf16 via async global->LDS DMA
#pragma unroll
        for (int rep = 0; rep < 8; rep++) {
            int idx = tid + rep * 256;           // 0..2047 chunks of 8 bf16
            int pi  = idx >> 5;                  // pixel in tile
            int ch  = (idx & 31) * 8;            // channel chunk
            int p   = m0 + pi;
            int b   = p / SP, s = p % SP;
            int yy  = s / W_ + ky - 1, xc = s % W_ + kx - 1;
            unsigned ldsa = (unsigned)(uintptr_t)&As[pi * C_ + ch];
            if (yy >= 0 && yy < H_ && xc >= 0 && xc < W_) {
                unsigned long long ga = (unsigned long long)(uintptr_t)
                    &xb[(((size_t)b * SP + yy * W_ + xc)) * C_ + ch];
                asm volatile("global_load_async_to_lds_b128 %0, %1, off"
                             :: "v"(ldsa), "v"(ga) : "memory");
            } else {
                u16x8 z = {0,0,0,0,0,0,0,0};
                *(u16x8*)&As[pi * C_ + ch] = z;
            }
        }
        asm volatile("s_wait_asynccnt 0x0" ::: "memory");
        __syncthreads();

        // hint the next tap's B panels toward L2
        if (tap < 8)
            __builtin_prefetch(&bp[(((size_t)((tap + 1) * 16)) * C_ + n0) * 16], 0, 1);

        const unsigned short* arow = &As[(mi * 16 + r) * C_];
        int col0 = n0 + np * 64 + r;
        for (int kc = 0; kc < C_; kc += 32) {
            FragU a;
            a.h[0] = *(const u16x8*)&arow[kc + h * 16];
            a.h[1] = *(const u16x8*)&arow[kc + h * 16 + 8];
            int pan = ((tap * C_ + kc) >> 4) + h;
            const unsigned short* bbase = bp + (size_t)pan * C_ * 16;
#pragma unroll
            for (int t2 = 0; t2 < 4; t2++) {
                FragU bf;
                bf.h[0] = *(const u16x8*)&bbase[(col0 + t2 * 16) * 16];
                bf.h[1] = *(const u16x8*)&bbase[(col0 + t2 * 16) * 16 + 8];
                acc[t2] = wmma_bf16(a.v, bf.v, acc[t2]);
            }
        }
    }

    // epilogue: BN + SiLU, write bf16 into cat channels 128..383
    int pm = m0 + mi * 16 + h * 8;
#pragma unroll
    for (int t2 = 0; t2 < 4; t2++) {
        int n = n0 + np * 64 + t2 * 16 + r;
        float sc2 = inv[n], bi = bias[n];
        f32x8 a = acc[t2];
#pragma unroll
        for (int i = 0; i < 8; i++) {
            float v2 = siluf(a[i] * sc2 + bi);
            cat[(size_t)(pm + i) * CATC + 128 + n] = f2bf(v2);
        }
    }
}

// ---------------- 1x1 out conv 384->256 : GEMM via WMMA, A staged by TDM ----------------
// block tile: M=64 pixels x N=64 channels. One wave issues tensor_load_to_lds for the
// 64x384 bf16 A slab (2D tile of the concat tensor), all waves consume from LDS.
__global__ void k_out_conv(const unsigned short* __restrict__ cat,
                           const unsigned short* __restrict__ bp,
                           const float* __restrict__ inv, const float* __restrict__ bias,
                           float* __restrict__ out) {
    __shared__ __align__(16) unsigned short As2[64 * CATC];   // 48 KB
    int tid  = threadIdx.x;
    int wv   = tid >> 5, lane = tid & 31;
    int h    = lane >> 4, r = lane & 15;
    int mi   = wv & 3, np = wv >> 2;
    int m0   = blockIdx.x * 64;
    int n0   = blockIdx.y * 64;

    if (tid == 0) {
        unsigned ldsa = (unsigned)(uintptr_t)As2;
        unsigned long long ga =
            (unsigned long long)(uintptr_t)(cat + (size_t)m0 * CATC);
        // D# group0: count=1 | lds_addr | global_addr | type=2 (bits 127:126)
        u32x4 g0 = { 1u, ldsa, (unsigned)ga, (unsigned)(ga >> 32) | (2u << 30) };
        // D# group1: data_size=2B, tensor 384x64 (stride 384), tile 384x64, no pad/iter
        u32x8 g1 = { 0x00010000u,             // data_size=1 (2 bytes) at bits 17:16
                     (384u & 0xFFFF) << 16,   // tensor_dim0 low16 at bits 63:48
                     (64u  & 0xFFFF) << 16,   // dim0 hi16 | tensor_dim1 low16
                     384u << 16,              // dim1 hi16 | tile_dim0
                     64u,                     // tile_dim1 | tile_dim2=0
                     384u,                    // tensor_dim0_stride low32
                     0u, 0u };                // stride0 hi16 | stride1
        asm volatile("tensor_load_to_lds %0, %1" :: "s"(g0), "s"(g1) : "memory");
        __builtin_amdgcn_s_wait_tensorcnt(0);
    }
    __syncthreads();

    f32x8 acc0 = {0,0,0,0,0,0,0,0};
    f32x8 acc1 = {0,0,0,0,0,0,0,0};

    const unsigned short* arow = &As2[(mi * 16 + r) * CATC];
    int col0 = n0 + np * 32 + r;
#pragma unroll
    for (int kc = 0; kc < CATC; kc += 32) {
        FragU a;
        a.h[0] = *(const u16x8*)&arow[kc + h * 16];
        a.h[1] = *(const u16x8*)&arow[kc + h * 16 + 8];
        int pan = (kc >> 4) + h;
        const unsigned short* bbase = bp + (size_t)pan * C_ * 16;
        FragU b0, b1;
        b0.h[0] = *(const u16x8*)&bbase[col0 * 16];
        b0.h[1] = *(const u16x8*)&bbase[col0 * 16 + 8];
        b1.h[0] = *(const u16x8*)&bbase[(col0 + 16) * 16];
        b1.h[1] = *(const u16x8*)&bbase[(col0 + 16) * 16 + 8];
        acc0 = wmma_bf16(a.v, b0.v, acc0);
        acc1 = wmma_bf16(a.v, b1.v, acc1);
    }

    // epilogue: BN + SiLU, scatter fp32 to NCHW d_out
    int pm = m0 + mi * 16 + h * 8;
#pragma unroll
    for (int tile = 0; tile < 2; tile++) {
        int n = n0 + np * 32 + tile * 16 + r;
        float sc2 = inv[n], bi = bias[n];
        f32x8 a = tile ? acc1 : acc0;
#pragma unroll
        for (int i = 0; i < 8; i++) {
            int p = pm + i;
            int b = p / SP, s = p % SP;
            out[((size_t)(b * C_ + n)) * SP + s] = siluf(a[i] * sc2 + bi);
        }
    }
}

// ---------------- host launcher ----------------
extern "C" void kernel_launch(void* const* d_in, const int* in_sizes, int n_in,
                              void* d_out, int out_size, void* d_ws, size_t ws_size,
                              hipStream_t stream) {
    (void)in_sizes; (void)n_in; (void)out_size; (void)ws_size;
    const float* x      = (const float*)d_in[0];
    const float* qkv_w  = (const float*)d_in[1];
    const float* qkv_g  = (const float*)d_in[2];
    const float* qkv_b  = (const float*)d_in[3];
    const float* qkv_m  = (const float*)d_in[4];
    const float* qkv_v  = (const float*)d_in[5];
    const float* hid_w  = (const float*)d_in[6];
    const float* hid_g  = (const float*)d_in[7];
    const float* hid_b  = (const float*)d_in[8];
    const float* hid_m  = (const float*)d_in[9];
    const float* hid_v  = (const float*)d_in[10];
    const float* out_w  = (const float*)d_in[11];
    const float* out_g  = (const float*)d_in[12];
    const float* out_b  = (const float*)d_in[13];
    const float* out_m  = (const float*)d_in[14];
    const float* out_v  = (const float*)d_in[15];

    char* ws = (char*)d_ws;
    size_t off = 0;
    auto take = [&](size_t n) -> char* {
        char* p = ws + off;
        off = (off + n + 255) & ~(size_t)255;
        return p;
    };
    unsigned short* xb     = (unsigned short*)take((size_t)NPIX * C_ * 2);   // 52.4 MB
    unsigned short* cat    = (unsigned short*)take((size_t)NPIX * CATC * 2); // 78.6 MB
    unsigned short* hidBp  = (unsigned short*)take((size_t)144 * C_ * 16 * 2);
    unsigned short* outBp  = (unsigned short*)take((size_t)24 * C_ * 16 * 2);
    float* q_inv  = (float*)take(QKVO * 4);
    float* q_bias = (float*)take(QKVO * 4);
    float* h_inv  = (float*)take(C_ * 4);
    float* h_bias = (float*)take(C_ * 4);
    float* o_inv  = (float*)take(C_ * 4);
    float* o_bias = (float*)take(C_ * 4);

    k_x_to_nhwc<<<dim3(SP / 32, C_ / 32, B_), 256, 0, stream>>>(x, xb);
    k_fold_bn<<<1, 256, 0, stream>>>(qkv_g, qkv_b, qkv_m, qkv_v,
                                     hid_g, hid_b, hid_m, hid_v,
                                     out_g, out_b, out_m, out_v,
                                     q_inv, q_bias, h_inv, h_bias, o_inv, o_bias);
    k_pack_hid<<<(C_ * C_ * 9) / 256, 256, 0, stream>>>(hid_w, hidBp);
    k_pack_out<<<(C_ * CATC) / 256, 256, 0, stream>>>(out_w, outBp);

    k_qkv_attn<<<NPIX / 16, 256, 0, stream>>>(x, qkv_w, q_inv, q_bias, cat);
    k_hid_conv<<<dim3(NPIX / 64, C_ / 128), 256, 0, stream>>>(xb, hidBp, h_inv, h_bias, cat);
    k_out_conv<<<dim3(NPIX / 64, C_ / 64), 256, 0, stream>>>(cat, outBp, o_inv, o_bias,
                                                             (float*)d_out);
}
